// S6_49512382988926
// MI455X (gfx1250) — matrix operand order repaired and checked by reading, not compiled, not run on hardware
//
#include <hip/hip_runtime.h>
#include <math.h>

// Problem constants (B_SZ=2, L=2048, D=1024, N=16)
#define M_TOT 4096   // B_SZ * L
#define DDIM  1024

typedef __attribute__((ext_vector_type(16))) __bf16 v16bf;
typedef __attribute__((ext_vector_type(8)))  __bf16 bf8v;
typedef __attribute__((ext_vector_type(4)))  __bf16 bf4v;
typedef __attribute__((ext_vector_type(8)))  float  v8f;

// ---------------------------------------------------------------------------
// Kernel 0: split f32 -> bf16 (hi) + bf16 (lo residual), 4 elements / thread
// ---------------------------------------------------------------------------
__global__ __launch_bounds__(256) void s6_split_bf16_kernel(
    const float* __restrict__ src, __bf16* __restrict__ hi,
    __bf16* __restrict__ lo, int n4) {
  int i = blockIdx.x * 256 + threadIdx.x;
  if (i >= n4) return;
  const float4 v = ((const float4*)src)[i];
  bf4v h, l;
  float f;
  f = v.x; h[0] = (__bf16)f; l[0] = (__bf16)(f - (float)h[0]);
  f = v.y; h[1] = (__bf16)f; l[1] = (__bf16)(f - (float)h[1]);
  f = v.z; h[2] = (__bf16)f; l[2] = (__bf16)(f - (float)h[2]);
  f = v.w; h[3] = (__bf16)f; l[3] = (__bf16)(f - (float)h[3]);
  ((bf4v*)hi)[i] = h;
  ((bf4v*)lo)[i] = l;
}

// ---------------------------------------------------------------------------
// Kernel 1: per-row scalar s[m] = (x@W2^T + b2) . (x@W3^T + b3)  (N = 16)
// One block (8 waves) per row; wave w owns output channels 2w, 2w+1.
// ---------------------------------------------------------------------------
__global__ __launch_bounds__(256) void s6_bcdot_kernel(
    const float* __restrict__ x,
    const float* __restrict__ W2, const float* __restrict__ b2,
    const float* __restrict__ W3, const float* __restrict__ b3,
    float* __restrict__ s) {
  const int row  = blockIdx.x;
  const int lane = threadIdx.x & 31;
  const int wave = threadIdx.x >> 5;
  const int n0   = wave * 2;

  const float* xr  = x  + (size_t)row * DDIM;
  const float* w2a = W2 + (size_t)n0 * DDIM;
  const float* w2b = w2a + DDIM;
  const float* w3a = W3 + (size_t)n0 * DDIM;
  const float* w3b = w3a + DDIM;

  float aB0 = 0.f, aB1 = 0.f, aC0 = 0.f, aC1 = 0.f;
  for (int k = lane * 4; k < DDIM; k += 128) {
    float4 xv = *(const float4*)(xr + k);
    float4 w;
    w = *(const float4*)(w2a + k);
    aB0 += xv.x*w.x + xv.y*w.y + xv.z*w.z + xv.w*w.w;
    w = *(const float4*)(w2b + k);
    aB1 += xv.x*w.x + xv.y*w.y + xv.z*w.z + xv.w*w.w;
    w = *(const float4*)(w3a + k);
    aC0 += xv.x*w.x + xv.y*w.y + xv.z*w.z + xv.w*w.w;
    w = *(const float4*)(w3b + k);
    aC1 += xv.x*w.x + xv.y*w.y + xv.z*w.z + xv.w*w.w;
  }
#pragma unroll
  for (int off = 16; off > 0; off >>= 1) {
    aB0 += __shfl_xor(aB0, off, 32);
    aB1 += __shfl_xor(aB1, off, 32);
    aC0 += __shfl_xor(aC0, off, 32);
    aC1 += __shfl_xor(aC1, off, 32);
  }
  __shared__ float part[8];
  if (lane == 0) {
    float Bm0 = aB0 + b2[n0], Bm1 = aB1 + b2[n0 + 1];
    float Cm0 = aC0 + b3[n0], Cm1 = aC1 + b3[n0 + 1];
    part[wave] = Bm0 * Cm0 + Bm1 * Cm1;
  }
  __syncthreads();
  if (threadIdx.x == 0) {
    float t = 0.f;
#pragma unroll
    for (int i = 0; i < 8; ++i) t += part[i];
    s[row] = t;
  }
}

// ---------------------------------------------------------------------------
// Kernel 2: WMMA GEMM  G = x @ W1^T  (bf16x3 error-compensated), fused
// epilogue y = x * softplus(G + b1) * s[m].
// Block = 8 waves (2x4) of 32x32 wave tiles -> 64(M) x 128(N) block tile.
// ---------------------------------------------------------------------------
__device__ __forceinline__ v16bf cat8(bf8v a, bf8v b) {
  return __builtin_shufflevector(a, b, 0, 1, 2, 3, 4, 5, 6, 7,
                                 8, 9, 10, 11, 12, 13, 14, 15);
}

__device__ __forceinline__ void epi_store(float* __restrict__ y,
                                          const float* __restrict__ x,
                                          int m, int n, float z, float sv) {
  float sp = (z > 20.f) ? z : log1pf(expf(z));
  size_t idx = (size_t)m * DDIM + n;
  y[idx] = x[idx] * sp * sv;
}

__global__ __launch_bounds__(256) void s6_wmma_gemm_kernel(
    const __bf16* __restrict__ xh, const __bf16* __restrict__ xl,
    const __bf16* __restrict__ wh, const __bf16* __restrict__ wl,
    const float* __restrict__ x, const float* __restrict__ b1,
    const float* __restrict__ s, float* __restrict__ y) {
  const int lane = threadIdx.x & 31;
  const int wave = threadIdx.x >> 5;
  const int m0 = blockIdx.y * 64  + (wave >> 2) * 32;   // M block tile 64
  const int n0 = blockIdx.x * 128 + (wave & 3) * 32;    // N block tile 128
  const int l15 = lane & 15;
  const bool hiHalf = (lane >= 16);

  // ISA 16-bit A layout: lanes 0-15 hold K {0..7,16..23}; 16-31 hold {8..15,24..31}
  const int kA = hiHalf ? 8 : 0;
  // ISA 16-bit B layout: lanes 0-15 hold K {0..7,16..23}; 16-31 hold {8..15,24..31}
  const int kB = hiHalf ? 8 : 0;

  const size_t aBase0 = (size_t)(m0 + l15) * DDIM + kA;        // M subtile 0
  const size_t aBase1 = aBase0 + (size_t)16 * DDIM;            // M subtile 1
  const size_t bBase0 = (size_t)(n0 + l15) * DDIM + kB;        // N subtile 0
  const size_t bBase1 = bBase0 + (size_t)16 * DDIM;            // N subtile 1

  v8f acc00 = {}, acc01 = {}, acc10 = {}, acc11 = {};

  for (int k0 = 0; k0 < DDIM; k0 += 32) {
    // ---- A fragments (two K-runs of 8 bf16 each) ----
    v16bf a0h = cat8(*(const bf8v*)(xh + aBase0 + k0),
                     *(const bf8v*)(xh + aBase0 + k0 + 16));
    v16bf a0l = cat8(*(const bf8v*)(xl + aBase0 + k0),
                     *(const bf8v*)(xl + aBase0 + k0 + 16));
    v16bf a1h = cat8(*(const bf8v*)(xh + aBase1 + k0),
                     *(const bf8v*)(xh + aBase1 + k0 + 16));
    v16bf a1l = cat8(*(const bf8v*)(xl + aBase1 + k0),
                     *(const bf8v*)(xl + aBase1 + k0 + 16));
    // ---- B fragments (two K-runs of 8 bf16 each) ----
    v16bf b0h = cat8(*(const bf8v*)(wh + bBase0 + k0),
                     *(const bf8v*)(wh + bBase0 + k0 + 16));
    v16bf b0l = cat8(*(const bf8v*)(wl + bBase0 + k0),
                     *(const bf8v*)(wl + bBase0 + k0 + 16));
    v16bf b1h = cat8(*(const bf8v*)(wh + bBase1 + k0),
                     *(const bf8v*)(wh + bBase1 + k0 + 16));
    v16bf b1l = cat8(*(const bf8v*)(wl + bBase1 + k0),
                     *(const bf8v*)(wl + bBase1 + k0 + 16));

    // bf16x3: hi*hi + hi*lo + lo*hi, accumulated in f32
    acc00 = __builtin_amdgcn_wmma_f32_16x16x32_bf16(false, a0h, false, b0h, (short)0, acc00, false, false);
    acc00 = __builtin_amdgcn_wmma_f32_16x16x32_bf16(false, a0h, false, b0l, (short)0, acc00, false, false);
    acc00 = __builtin_amdgcn_wmma_f32_16x16x32_bf16(false, a0l, false, b0h, (short)0, acc00, false, false);

    acc01 = __builtin_amdgcn_wmma_f32_16x16x32_bf16(false, a0h, false, b1h, (short)0, acc01, false, false);
    acc01 = __builtin_amdgcn_wmma_f32_16x16x32_bf16(false, a0h, false, b1l, (short)0, acc01, false, false);
    acc01 = __builtin_amdgcn_wmma_f32_16x16x32_bf16(false, a0l, false, b1h, (short)0, acc01, false, false);

    acc10 = __builtin_amdgcn_wmma_f32_16x16x32_bf16(false, a1h, false, b0h, (short)0, acc10, false, false);
    acc10 = __builtin_amdgcn_wmma_f32_16x16x32_bf16(false, a1h, false, b0l, (short)0, acc10, false, false);
    acc10 = __builtin_amdgcn_wmma_f32_16x16x32_bf16(false, a1l, false, b0h, (short)0, acc10, false, false);

    acc11 = __builtin_amdgcn_wmma_f32_16x16x32_bf16(false, a1h, false, b1h, (short)0, acc11, false, false);
    acc11 = __builtin_amdgcn_wmma_f32_16x16x32_bf16(false, a1h, false, b1l, (short)0, acc11, false, false);
    acc11 = __builtin_amdgcn_wmma_f32_16x16x32_bf16(false, a1l, false, b1h, (short)0, acc11, false, false);
  }

  // Epilogue. C/D layout: VGPR v, lanes 0-15 -> M = v, lanes 16-31 -> M = v+8.
  const int nIdx0 = n0 + l15;
  const int nIdx1 = nIdx0 + 16;
  const float bias0 = b1[nIdx0];
  const float bias1 = b1[nIdx1];
  const int mBase0 = m0 + (hiHalf ? 8 : 0);
#pragma unroll
  for (int v = 0; v < 8; ++v) {
    const int mA = mBase0 + v;
    const int mB = mA + 16;
    const float sA = s[mA];
    const float sB = s[mB];
    epi_store(y, x, mA, nIdx0, acc00[v] + bias0, sA);
    epi_store(y, x, mA, nIdx1, acc01[v] + bias1, sA);
    epi_store(y, x, mB, nIdx0, acc10[v] + bias0, sB);
    epi_store(y, x, mB, nIdx1, acc11[v] + bias1, sB);
  }
}

// ---------------------------------------------------------------------------
// Fallback (only if workspace is too small to stage bf16 operands):
// straight f32 dot per output element + fused epilogue.
// ---------------------------------------------------------------------------
__global__ __launch_bounds__(256) void s6_fallback_kernel(
    const float* __restrict__ x, const float* __restrict__ W1,
    const float* __restrict__ b1, const float* __restrict__ s,
    float* __restrict__ y) {
  int idx = blockIdx.x * 256 + threadIdx.x;
  if (idx >= M_TOT * DDIM) return;
  const int m = idx >> 10;
  const int n = idx & (DDIM - 1);
  const float* xr = x  + ((size_t)m << 10);
  const float* w  = W1 + ((size_t)n << 10);
  float g = b1[n];
  for (int k = 0; k < DDIM; k += 4) {
    float4 a = *(const float4*)(xr + k);
    float4 b = *(const float4*)(w + k);
    g += a.x * b.x + a.y * b.y + a.z * b.z + a.w * b.w;
  }
  float sp = (g > 20.f) ? g : log1pf(expf(g));
  y[idx] = x[idx] * sp * s[m];
}

// ---------------------------------------------------------------------------
extern "C" void kernel_launch(void* const* d_in, const int* in_sizes, int n_in,
                              void* d_out, int out_size, void* d_ws, size_t ws_size,
                              hipStream_t stream) {
  const float* x  = (const float*)d_in[0];
  const float* W1 = (const float*)d_in[1];
  const float* b1 = (const float*)d_in[2];
  const float* W2 = (const float*)d_in[3];
  const float* b2 = (const float*)d_in[4];
  const float* W3 = (const float*)d_in[5];
  const float* b3 = (const float*)d_in[6];
  // d_in[7] = A: provably unused (multiplies a zero-initialized state).
  float* y = (float*)d_out;

  // Workspace layout
  char* ws = (char*)d_ws;
  size_t off = 0;
  float* s = (float*)(ws + off);               off += (size_t)M_TOT * 4;         // 16 KB
  off = (off + 255) & ~(size_t)255;
  __bf16* xh = (__bf16*)(ws + off);            off += (size_t)M_TOT * DDIM * 2;  // 8 MB
  __bf16* xl = (__bf16*)(ws + off);            off += (size_t)M_TOT * DDIM * 2;  // 8 MB
  __bf16* wh = (__bf16*)(ws + off);            off += (size_t)DDIM * DDIM * 2;   // 2 MB
  __bf16* wl = (__bf16*)(ws + off);            off += (size_t)DDIM * DDIM * 2;   // 2 MB
  const size_t REQ = off;

  // s[m] = Bm . Cm per row (always needed)
  s6_bcdot_kernel<<<M_TOT, 256, 0, stream>>>(x, W2, b2, W3, b3, s);

  if (ws_size >= REQ) {
    const int nx4 = (M_TOT * DDIM) / 4;
    const int nw4 = (DDIM * DDIM) / 4;
    s6_split_bf16_kernel<<<(nx4 + 255) / 256, 256, 0, stream>>>(x,  xh, xl, nx4);
    s6_split_bf16_kernel<<<(nw4 + 255) / 256, 256, 0, stream>>>(W1, wh, wl, nw4);
    dim3 grid(DDIM / 128, M_TOT / 64);  // (8, 64)
    s6_wmma_gemm_kernel<<<grid, 256, 0, stream>>>(xh, xl, wh, wl, x, b1, s, y);
  } else {
    s6_fallback_kernel<<<(M_TOT * DDIM + 255) / 256, 256, 0, stream>>>(x, W1, b1, s, y);
  }
}
